// MPNNLayer_64424509440353
// MI455X (gfx1250) — matrix operand order repaired and checked by reading, not compiled, not run on hardware
//
#include <hip/hip_runtime.h>

// ---------------------------------------------------------------------------
// MPNN layer for MI455X (gfx1250), bf16 WMMA path.
//   u_ef = MLP_e([nf[src] | nf[dst] | ef])          (96 -> 64 -> 64 -> 32, ReLU)
//   agg  = segment_sum(u_ef, dst)                   (fused hw f32 atomics)
//   u_nf = MLP_n([nf | agg])                        (64 -> 64 -> 64 -> 32, ReLU)
// d_out = [u_nf (N*32) | u_ef (E*32)], all f32.
//
// Memory plan: ef read (205MB) and u_ef write (205MB) are streamed with
// non-temporal hints; nf (12.8MB) and agg (12.8MB) stay L2-resident for the
// random gathers / scatter atomics. ~620MB @ 23.3TB/s ≈ 27us roof; bf16 WMMA
// keeps the 41 GFLOP of MLP work under that roof.
// ---------------------------------------------------------------------------

typedef __attribute__((ext_vector_type(16))) __bf16 v16bf;
typedef __attribute__((ext_vector_type(8)))  __bf16 v8bf;
typedef __attribute__((ext_vector_type(8)))  float  v8f;
typedef __attribute__((ext_vector_type(4)))  float  v4f;   // native vector for NT builtins

#define WMMA_BF16(A, B, C) \
  __builtin_amdgcn_wmma_f32_16x16x32_bf16(false, (A), false, (B), (short)0, (C), false, false)

#define NODE_F 32
#define HID    64
#define EOUT   32
#define HSTR   72   // padded LDS row stride (bf16 elems); 72*2=144B, 16B aligned

// ---- A-fragment (16x32, M=row, K=32) loaded from a contiguous 32-float row.
// ISA layout: lanes 0-15 hold row M=lane, K = {0..7, 16..23};
//             lanes 16-31 hold row M=lane-16, K = {8..15, 24..31}.
template <bool NT>
__device__ __forceinline__ v16bf load_a_row32(const float* __restrict__ row, int half) {
  const v4f* p = (const v4f*)row;
  v4f q0, q1, q2, q3;
  if (NT) {
    q0 = __builtin_nontemporal_load(p + half * 2 + 0);
    q1 = __builtin_nontemporal_load(p + half * 2 + 1);
    q2 = __builtin_nontemporal_load(p + half * 2 + 4);
    q3 = __builtin_nontemporal_load(p + half * 2 + 5);
  } else {
    q0 = p[half * 2 + 0];
    q1 = p[half * 2 + 1];
    q2 = p[half * 2 + 4];
    q3 = p[half * 2 + 5];
  }
  v16bf r;
  r[0]  = (__bf16)q0.x; r[1]  = (__bf16)q0.y; r[2]  = (__bf16)q0.z; r[3]  = (__bf16)q0.w;
  r[4]  = (__bf16)q1.x; r[5]  = (__bf16)q1.y; r[6]  = (__bf16)q1.z; r[7]  = (__bf16)q1.w;
  r[8]  = (__bf16)q2.x; r[9]  = (__bf16)q2.y; r[10] = (__bf16)q2.z; r[11] = (__bf16)q2.w;
  r[12] = (__bf16)q3.x; r[13] = (__bf16)q3.y; r[14] = (__bf16)q3.z; r[15] = (__bf16)q3.w;
  return r;
}

// ---- B-fragment (32x16, K x N) from row-major W[fan_in][fan_out] (f32 -> bf16).
// B layout: lanes 0-15 hold col N=lane, K = kc*32 + {0..7, 16..23};
//           lanes 16-31 hold col N=lane-16, K = kc*32 + {8..15, 24..31}.
__device__ __forceinline__ v16bf load_b(const float* __restrict__ W, int fanout,
                                        int kc, int nc, int lane) {
  int col   = nc * 16 + (lane & 15);
  int kbase = kc * 32 + ((lane & 16) ? 8 : 0);
  const float* p = W + (size_t)kbase * fanout + col;
  v16bf r;
#pragma unroll
  for (int i = 0; i < 16; ++i) r[i] = (__bf16)p[(size_t)(i < 8 ? i : i + 8) * fanout];
  return r;
}

__device__ __forceinline__ v8f splat8(float b) {
  v8f c;
#pragma unroll
  for (int i = 0; i < 8; ++i) c[i] = b;
  return c;
}

// ---- C/D (16x16 f32) fragment -> LDS tile [16][HSTR] bf16, with ReLU.
// C layout: VGPR v: lanes 0-15 -> (M=v, N=lane); lanes 16-31 -> (M=v+8, N=lane-16).
__device__ __forceinline__ void frag_to_lds(__bf16* __restrict__ h, v8f acc, int nc, int lane) {
  int col   = nc * 16 + (lane & 15);
  int rbase = (lane & 16) ? 8 : 0;
#pragma unroll
  for (int v = 0; v < 8; ++v)
    h[(rbase + v) * HSTR + col] = (__bf16)fmaxf(acc[v], 0.0f);
}

// ---- LDS tile -> A-fragment for the next layer's K-chunk kc.
__device__ __forceinline__ v16bf lds_to_afrag(const __bf16* __restrict__ h, int kc, int lane) {
  int m  = lane & 15;
  int kb = kc * 32 + ((lane & 16) ? 8 : 0);
  const __bf16* row = h + m * HSTR;
  v8bf lo = *(const v8bf*)(row + kb);        // K = kb .. kb+7   (16B aligned)
  v8bf hi = *(const v8bf*)(row + kb + 16);   // K = kb+16 .. kb+23
  v16bf r;
#pragma unroll
  for (int i = 0; i < 8; ++i) { r[i] = lo[i]; r[8 + i] = hi[i]; }
  return r;
}

// ===========================================================================
// Edge kernel: 16-edge tiles per wave; 24 WMMAs per tile; fused scatter-add.
// ===========================================================================
__global__ __launch_bounds__(128) void mpnn_edge_kernel(
    const float* __restrict__ nf, const float* __restrict__ ef,
    const int* __restrict__ src, const int* __restrict__ dst,
    const float* __restrict__ W1, const float* __restrict__ b1,
    const float* __restrict__ W2, const float* __restrict__ b2,
    const float* __restrict__ W3, const float* __restrict__ b3,
    float* __restrict__ u_ef, float* __restrict__ agg, int nTiles)
{
  __shared__ __align__(16) __bf16 hbuf[4][16 * HSTR];
  const int lane = threadIdx.x & 31;
  const int w    = threadIdx.x >> 5;
  __bf16* h = hbuf[w];
  const int wave   = blockIdx.x * 4 + w;
  const int nWaves = gridDim.x * 4;

  // Weight fragments, held in VGPRs for the whole persistent loop.
  v16bf B1[3][4], B2[2][4], B3[2][2];
#pragma unroll
  for (int kc = 0; kc < 3; ++kc)
#pragma unroll
    for (int nc = 0; nc < 4; ++nc) B1[kc][nc] = load_b(W1, HID, kc, nc, lane);
#pragma unroll
  for (int kc = 0; kc < 2; ++kc)
#pragma unroll
    for (int nc = 0; nc < 4; ++nc) B2[kc][nc] = load_b(W2, HID, kc, nc, lane);
#pragma unroll
  for (int kc = 0; kc < 2; ++kc)
#pragma unroll
    for (int nc = 0; nc < 2; ++nc) B3[kc][nc] = load_b(W3, EOUT, kc, nc, lane);

  float b1v[4], b2v[4], b3v[2];
#pragma unroll
  for (int n = 0; n < 4; ++n) { b1v[n] = b1[n * 16 + (lane & 15)]; b2v[n] = b2[n * 16 + (lane & 15)]; }
#pragma unroll
  for (int n = 0; n < 2; ++n) b3v[n] = b3[n * 16 + (lane & 15)];

  const int m    = lane & 15;
  const int half = (lane >> 4) & 1;

  for (int t = wave; t < nTiles; t += nWaves) {
    const int e = t * 16 + m;
    const int s = src[e];
    const int d = dst[e];

    v16bf A0 = load_a_row32<false>(nf + (size_t)s * NODE_F, half);  // K 0..31  : nf[src] (L2-resident)
    v16bf A1 = load_a_row32<false>(nf + (size_t)d * NODE_F, half);  // K 32..63 : nf[dst]
    v16bf A2 = load_a_row32<true >(ef + (size_t)e * NODE_F, half);  // K 64..95 : ef (stream, NT)

    // Layer 1: [16x96] @ [96x64]
    v8f acc[4];
#pragma unroll
    for (int n = 0; n < 4; ++n) {
      acc[n] = splat8(b1v[n]);
      acc[n] = WMMA_BF16(A0, B1[0][n], acc[n]);
      acc[n] = WMMA_BF16(A1, B1[1][n], acc[n]);
      acc[n] = WMMA_BF16(A2, B1[2][n], acc[n]);
    }
#pragma unroll
    for (int n = 0; n < 4; ++n) frag_to_lds(h, acc[n], n, lane);
    v16bf H0 = lds_to_afrag(h, 0, lane);
    v16bf H1 = lds_to_afrag(h, 1, lane);

    // Layer 2: [16x64] @ [64x64]
#pragma unroll
    for (int n = 0; n < 4; ++n) {
      acc[n] = splat8(b2v[n]);
      acc[n] = WMMA_BF16(H0, B2[0][n], acc[n]);
      acc[n] = WMMA_BF16(H1, B2[1][n], acc[n]);
    }
#pragma unroll
    for (int n = 0; n < 4; ++n) frag_to_lds(h, acc[n], n, lane);
    H0 = lds_to_afrag(h, 0, lane);
    H1 = lds_to_afrag(h, 1, lane);

    // Layer 3: [16x64] @ [64x32]
    v8f out[2];
#pragma unroll
    for (int n = 0; n < 2; ++n) {
      out[n] = splat8(b3v[n]);
      out[n] = WMMA_BF16(H0, B3[0][n], out[n]);
      out[n] = WMMA_BF16(H1, B3[1][n], out[n]);
    }

    // Store u_ef (ReLU, non-temporal stream) + fused segment-sum via hardware
    // f32 atomics into the L2-resident aggregator.
    const int rbase = t * 16 + half * 8;
#pragma unroll
    for (int n = 0; n < 2; ++n) {
      const int col = n * 16 + m;
#pragma unroll
      for (int v = 0; v < 8; ++v) {
        const float val = fmaxf(out[n][v], 0.0f);
        const int er = rbase + v;                  // edge row of this element
        __builtin_nontemporal_store(val, &u_ef[(size_t)er * EOUT + col]);
        const int dr = dst[er];                    // broadcast-ish load, L2 hit
        unsafeAtomicAdd(&agg[(size_t)dr * EOUT + col], val);
      }
    }
  }
}

// ===========================================================================
// Node kernel: 16-node tiles; input [nf | agg] (K=64); 20 WMMAs per tile.
// ===========================================================================
__global__ __launch_bounds__(128) void mpnn_node_kernel(
    const float* __restrict__ nf, const float* __restrict__ agg,
    const float* __restrict__ W1, const float* __restrict__ b1,
    const float* __restrict__ W2, const float* __restrict__ b2,
    const float* __restrict__ W3, const float* __restrict__ b3,
    float* __restrict__ u_nf, int nTiles)
{
  __shared__ __align__(16) __bf16 hbuf[4][16 * HSTR];
  const int lane = threadIdx.x & 31;
  const int w    = threadIdx.x >> 5;
  __bf16* h = hbuf[w];
  const int wave   = blockIdx.x * 4 + w;
  const int nWaves = gridDim.x * 4;

  v16bf B1[2][4], B2[2][4], B3[2][2];
#pragma unroll
  for (int kc = 0; kc < 2; ++kc)
#pragma unroll
    for (int nc = 0; nc < 4; ++nc) B1[kc][nc] = load_b(W1, HID, kc, nc, lane);
#pragma unroll
  for (int kc = 0; kc < 2; ++kc)
#pragma unroll
    for (int nc = 0; nc < 4; ++nc) B2[kc][nc] = load_b(W2, HID, kc, nc, lane);
#pragma unroll
  for (int kc = 0; kc < 2; ++kc)
#pragma unroll
    for (int nc = 0; nc < 2; ++nc) B3[kc][nc] = load_b(W3, NODE_F, kc, nc, lane);

  float b1v[4], b2v[4], b3v[2];
#pragma unroll
  for (int n = 0; n < 4; ++n) { b1v[n] = b1[n * 16 + (lane & 15)]; b2v[n] = b2[n * 16 + (lane & 15)]; }
#pragma unroll
  for (int n = 0; n < 2; ++n) b3v[n] = b3[n * 16 + (lane & 15)];

  const int m    = lane & 15;
  const int half = (lane >> 4) & 1;

  for (int t = wave; t < nTiles; t += nWaves) {
    const int r = t * 16 + m;
    v16bf A0 = load_a_row32<false>(nf  + (size_t)r * NODE_F, half);  // K 0..31  : nf
    v16bf A1 = load_a_row32<false>(agg + (size_t)r * NODE_F, half);  // K 32..63 : agg

    v8f acc[4];
#pragma unroll
    for (int n = 0; n < 4; ++n) {
      acc[n] = splat8(b1v[n]);
      acc[n] = WMMA_BF16(A0, B1[0][n], acc[n]);
      acc[n] = WMMA_BF16(A1, B1[1][n], acc[n]);
    }
#pragma unroll
    for (int n = 0; n < 4; ++n) frag_to_lds(h, acc[n], n, lane);
    v16bf H0 = lds_to_afrag(h, 0, lane);
    v16bf H1 = lds_to_afrag(h, 1, lane);

#pragma unroll
    for (int n = 0; n < 4; ++n) {
      acc[n] = splat8(b2v[n]);
      acc[n] = WMMA_BF16(H0, B2[0][n], acc[n]);
      acc[n] = WMMA_BF16(H1, B2[1][n], acc[n]);
    }
#pragma unroll
    for (int n = 0; n < 4; ++n) frag_to_lds(h, acc[n], n, lane);
    H0 = lds_to_afrag(h, 0, lane);
    H1 = lds_to_afrag(h, 1, lane);

    v8f out[2];
#pragma unroll
    for (int n = 0; n < 2; ++n) {
      out[n] = splat8(b3v[n]);
      out[n] = WMMA_BF16(H0, B3[0][n], out[n]);
      out[n] = WMMA_BF16(H1, B3[1][n], out[n]);
    }

    const int rb = t * 16 + half * 8;
#pragma unroll
    for (int n = 0; n < 2; ++n) {
      const int col = n * 16 + m;
#pragma unroll
      for (int v = 0; v < 8; ++v)
        __builtin_nontemporal_store(fmaxf(out[n][v], 0.0f),
                                    &u_nf[(size_t)(rb + v) * NODE_F + col]);
    }
  }
}

// ===========================================================================
extern "C" void kernel_launch(void* const* d_in, const int* in_sizes, int n_in,
                              void* d_out, int out_size, void* d_ws, size_t ws_size,
                              hipStream_t stream) {
  const float* nf  = (const float*)d_in[0];
  const float* ef  = (const float*)d_in[1];
  const int*   src = (const int*)d_in[2];
  const int*   dst = (const int*)d_in[3];
  const float* eW1 = (const float*)d_in[4];
  const float* eb1 = (const float*)d_in[5];
  const float* eW2 = (const float*)d_in[6];
  const float* eb2 = (const float*)d_in[7];
  const float* eW3 = (const float*)d_in[8];
  const float* eb3 = (const float*)d_in[9];
  const float* nW1 = (const float*)d_in[10];
  const float* nb1 = (const float*)d_in[11];
  const float* nW2 = (const float*)d_in[12];
  const float* nb2 = (const float*)d_in[13];
  const float* nW3 = (const float*)d_in[14];
  const float* nb3 = (const float*)d_in[15];

  const int N = in_sizes[0] / NODE_F;   // 100000
  const int E = in_sizes[1] / NODE_F;   // 1600000

  float* u_nf = (float*)d_out;
  float* u_ef = u_nf + (size_t)N * NODE_F;
  float* agg  = (float*)d_ws;           // [N, 32] f32 aggregator

  // Zero the aggregator (graph-capture-safe).
  (void)hipMemsetAsync(agg, 0, (size_t)N * NODE_F * sizeof(float), stream);

  // Edge MLP + fused scatter-add. E is a multiple of 16.
  mpnn_edge_kernel<<<2048, 128, 0, stream>>>(
      nf, ef, src, dst, eW1, eb1, eW2, eb2, eW3, eb3, u_ef, agg, E / 16);

  // Node MLP. N is a multiple of 16.
  mpnn_node_kernel<<<512, 128, 0, stream>>>(
      nf, agg, nW1, nb1, nW2, nb2, nW3, nb3, u_nf, N / 16);
}
